// _EfficientDensenetBottleneck_28475633172505
// MI455X (gfx1250) — matrix operand order repaired and checked by reading, not compiled, not run on hardware
//
#include <hip/hip_runtime.h>
#include <hip/hip_bf16.h>

typedef __attribute__((ext_vector_type(16))) __bf16        v16bf;
typedef __attribute__((ext_vector_type(8)))  float         v8f;
typedef __attribute__((ext_vector_type(4)))  float         v4f;
typedef __attribute__((ext_vector_type(4)))  unsigned int  v4u;

union BFrag { v4u q[2]; v16bf v; };

#define HW        3136        // 56*56
#define NIMG      32
#define CIN       512
#define COUT      128
#define NPIX      (NIMG * HW) // 100352
#define RT_STRIDE 40          // 32 k-elems + pad, bf16 units (80B rows, 16B aligned)

__device__ __forceinline__ unsigned short f2bf(float f) {
  unsigned int u = __float_as_uint(f);
  u += 0x7fffu + ((u >> 16) & 1u);       // round-to-nearest-even
  return (unsigned short)(u >> 16);
}

// ---------------- Kernel 1: per-channel batch stats -> scale/shift ----------
__global__ __launch_bounds__(256) void bn_stats(
    const float* __restrict__ x0, const float* __restrict__ x1,
    const float* __restrict__ x2, const float* __restrict__ x3,
    const float* __restrict__ gamma, const float* __restrict__ beta,
    float* __restrict__ scale, float* __restrict__ shift) {
  const int c   = blockIdx.x;            // 0..511 global channel
  const int tid = threadIdx.x;
  const float* xt = (c < 128) ? x0 : (c < 256) ? x1 : (c < 384) ? x2 : x3;
  const float* base = xt + (size_t)(c & 127) * HW;

  float s = 0.f, q = 0.f;
  for (int n = 0; n < NIMG; ++n) {
    const v4f* p = (const v4f*)(base + (size_t)n * 128 * HW);
    for (int i = tid; i < HW / 4; i += 256) {
      v4f v = p[i];
      s += v.x + v.y + v.z + v.w;
      q += v.x * v.x + v.y * v.y + v.z * v.z + v.w * v.w;
    }
  }
  __shared__ float rs[256], rq[256];
  rs[tid] = s; rq[tid] = q;
  __syncthreads();
  for (int off = 128; off > 0; off >>= 1) {
    if (tid < off) { rs[tid] += rs[tid + off]; rq[tid] += rq[tid + off]; }
    __syncthreads();
  }
  if (tid == 0) {
    const float invN = 1.f / (float)NPIX;
    float mean = rs[0] * invN;
    float var  = rq[0] * invN - mean * mean;
    float inv  = rsqrtf(var + 1e-5f);
    float sc   = gamma[c] * inv;
    scale[c] = sc;
    shift[c] = beta[c] - mean * sc;
  }
}

// ---------------- Kernel 2: conv weight fp32 -> bf16 [o][k] -----------------
__global__ __launch_bounds__(256) void pack_w(const float* __restrict__ w,
                                              unsigned short* __restrict__ wbf) {
  int i = blockIdx.x * 256 + threadIdx.x;  // 0..65535
  wbf[i] = f2bf(w[i]);
}

// ---------------- Kernel 3: fused BN-affine + ReLU + 1x1 conv (WMMA) -------
// Block: 256 threads = 8 waves. Tile: 128 out-ch x 64 pixels of one image.
// Wave w computes out-channels [16w, 16w+16) x 64 pixels = 4 D-tiles.
__global__ __launch_bounds__(256) void fused_gemm(
    const float* __restrict__ x0, const float* __restrict__ x1,
    const float* __restrict__ x2, const float* __restrict__ x3,
    const float* __restrict__ scale, const float* __restrict__ shift,
    const unsigned short* __restrict__ wbf, float* __restrict__ out) {
  __shared__ __align__(16) unsigned short Rt[64 * RT_STRIDE]; // [p][k] bf16
  __shared__ __align__(16) float s_scale[CIN];
  __shared__ __align__(16) float s_shift[CIN];

  const int tid   = threadIdx.x;
  const int lane  = tid & 31;
  const int wave  = tid >> 5;
  const int col   = lane & 15;   // = M row for A frag, N col for B/D frags
  const int halfl = lane >> 4;

  const int blk   = blockIdx.x;          // 1568 blocks
  const int nimg  = blk / 49;
  const int pbase = (blk % 49) * 64;

  // Staging ownership: one pixel x 8 consecutive channels per thread.
  const int sp  = tid & 63;              // pixel 0..63
  const int skg = tid >> 6;              // channel octet 0..3

  for (int i = tid; i < CIN; i += 256) { s_scale[i] = scale[i]; s_shift[i] = shift[i]; }

  v8f acc[4] = {};

  for (int kc = 0; kc < CIN; kc += 32) {
    // A fragment: W[16 o x 32 k] bf16 from global (L2-resident). Issue the
    // loads FIRST so their latency hides under the staging traffic below.
    // ISA A layout: lane m=col, half=halfl; VGPR v: k = (v/4)*16 + half*8 + (v%4)*2
    // -> two uint4 loads at dword offsets half*4 and half*4+8 within the row.
    BFrag a;
    const v4u* wrow = (const v4u*)(wbf + (size_t)(wave * 16 + col) * CIN + kc);
    a.q[0] = wrow[halfl];
    a.q[1] = wrow[halfl + 2];

    __syncthreads();   // previous chunk's B-fragment reads are done

    // Stage 32(k) x 64(p): normalize+ReLU+bf16, transposed [p][k] in LDS.
    {
      const int cg0 = kc + skg * 8;
      const float* xt  = (cg0 < 128) ? x0 : (cg0 < 256) ? x1 : (cg0 < 384) ? x2 : x3;
      const float* src = xt + ((size_t)(nimg * 128 + (cg0 & 127)) * HW) + pbase + sp;
      v4f s0 = *(const v4f*)&s_scale[cg0];
      v4f s1 = *(const v4f*)&s_scale[cg0 + 4];
      v4f h0 = *(const v4f*)&s_shift[cg0];
      v4f h1 = *(const v4f*)&s_shift[cg0 + 4];
      unsigned short us[8];
#pragma unroll
      for (int j = 0; j < 8; ++j) {
        float xv = src[(size_t)j * HW];
        float sc = (j < 4) ? s0[j] : s1[j - 4];
        float sh = (j < 4) ? h0[j] : h1[j - 4];
        us[j] = f2bf(fmaxf(fmaf(xv, sc, sh), 0.f));
      }
      v4u pk;
      pk.x = (unsigned int)us[0] | ((unsigned int)us[1] << 16);
      pk.y = (unsigned int)us[2] | ((unsigned int)us[3] << 16);
      pk.z = (unsigned int)us[4] | ((unsigned int)us[5] << 16);
      pk.w = (unsigned int)us[6] | ((unsigned int)us[7] << 16);
      *(v4u*)&Rt[sp * RT_STRIDE + skg * 8] = pk;   // one ds_store_b128

      // Pull next K-chunk toward the WGP (global_prefetch) while we compute.
      if (kc + 32 < CIN) {
        const int cgn = cg0 + 32;
        const float* xtn = (cgn < 128) ? x0 : (cgn < 256) ? x1 : (cgn < 384) ? x2 : x3;
        __builtin_prefetch(xtn + ((size_t)(nimg * 128 + (cgn & 127)) * HW) + pbase + sp, 0, 1);
      }
    }
    __syncthreads();

    // Load ALL four B fragments first (one ds clause), then issue the four
    // WMMAs back-to-back (disjoint accumulators; shared A; no ds waits
    // between matrix ops).
    BFrag b[4];
#pragma unroll
    for (int pt = 0; pt < 4; ++pt) {
      // B fragment: R[32 k x 16 p]; lane n=col, k = (e/8)*16 + half*8 + e%8 (matches A).
      const v4u* bp = (const v4u*)&Rt[(pt * 16 + col) * RT_STRIDE + halfl * 8];
      b[pt].q[0] = bp[0];
      b[pt].q[1] = bp[2];
    }
#pragma unroll
    for (int pt = 0; pt < 4; ++pt) {
      acc[pt] = __builtin_amdgcn_wmma_f32_16x16x32_bf16(
          false, a.v, false, b[pt].v, (short)0, acc[pt], false, false);
    }
  }

  // D layout: lane col = n, reg r -> m = r + 8*halfl.
  float* obase = out + (size_t)nimg * COUT * HW + pbase;
#pragma unroll
  for (int pt = 0; pt < 4; ++pt) {
#pragma unroll
    for (int r = 0; r < 8; ++r) {
      int oc = wave * 16 + halfl * 8 + r;
      obase[(size_t)oc * HW + pt * 16 + col] = acc[pt][r];
    }
  }
}

extern "C" void kernel_launch(void* const* d_in, const int* in_sizes, int n_in,
                              void* d_out, int out_size, void* d_ws, size_t ws_size,
                              hipStream_t stream) {
  const float* x0    = (const float*)d_in[0];
  const float* x1    = (const float*)d_in[1];
  const float* x2    = (const float*)d_in[2];
  const float* x3    = (const float*)d_in[3];
  const float* gamma = (const float*)d_in[4];
  const float* beta  = (const float*)d_in[5];
  const float* cw    = (const float*)d_in[6];

  float* scale         = (float*)d_ws;
  float* shift         = scale + CIN;
  unsigned short* wbf  = (unsigned short*)(shift + CIN);   // 128*512 bf16

  bn_stats<<<CIN, 256, 0, stream>>>(x0, x1, x2, x3, gamma, beta, scale, shift);
  pack_w<<<(COUT * CIN) / 256, 256, 0, stream>>>(cw, wbf);
  fused_gemm<<<NIMG * (HW / 64), 256, 0, stream>>>(x0, x1, x2, x3, scale, shift, wbf,
                                                   (float*)d_out);
}
